// SelfAttentionInteraction_2688649527805
// MI455X (gfx1250) — matrix operand order repaired and checked by reading, not compiled, not run on hardware
//
#include <hip/hip_runtime.h>
#include <cstdint>
#include <cstddef>

// ---------------------------------------------------------------------------
// SelfAttentionInteraction on gfx1250 (MI455X), bf16 WMMA pipeline with
// async global->LDS double buffering:
//   1) convert x, Wq, Wk, Wv -> bf16 workspace
//   2) f/g/h = x @ W^T           (gemm<bf16, B transposed>, store bf16)
//   3) scoresT[b,t,s] = g.f      (gemm, store f32)  -- transposed so the
//                                 query-axis softmax is a row softmax
//   4) betaT = softmax_rows(scoresT)  (store bf16)
//   5) out = gamma * betaT @ h + x    (gemm, fused epilogue, f32 out)
// ---------------------------------------------------------------------------

typedef __attribute__((ext_vector_type(16))) __bf16 v16bf;
typedef __attribute__((ext_vector_type(8)))  float  v8f;

#define BLK_M 128
#define BLK_N 64
#define BLK_K 32
#define KPAD  8
#define LDS_K (BLK_K + KPAD)

union FragBF16 { v16bf v; uint4 q[2]; };

// Async copy of 16 bytes (8 bf16) from global to LDS, tracked by ASYNCcnt.
// LDS aperture: wave-relative LDS byte offset lives in addr[31:0] (ISA 10.2),
// so truncating the generic pointer yields the instruction's LDS address.
__device__ __forceinline__ void async_copy_b128(const __bf16* gsrc, __bf16* ldst)
{
    uint32_t lds = (uint32_t)(size_t)ldst;
    asm volatile("global_load_async_to_lds_b128 %0, %1, off"
                 :
                 : "v"(lds), "v"(gsrc)
                 : "memory");
}

__global__ __launch_bounds__(256) void f32_to_bf16_kernel(
    const float* __restrict__ in, __bf16* __restrict__ out, size_t n)
{
    size_t i = (size_t)blockIdx.x * blockDim.x + threadIdx.x;
    size_t stride = (size_t)gridDim.x * blockDim.x;
    for (; i < n; i += stride) out[i] = (__bf16)in[i];
}

// C(MxN) = A(MxK) @ B ; B_TRANS ? B is (N x K) row-major : B is (K x N) row-major
// EPI: 0 = store f32 ; 1 = store bf16 ; 2 = out = gamma*acc + Xres (f32)
// M % 128 == 0, N % 64 == 0, K % 32 == 0 (true for all shapes here).
template <bool B_TRANS, int EPI>
__global__ __launch_bounds__(256) void gemm_wmma_bf16(
    const __bf16* __restrict__ A, const __bf16* __restrict__ B,
    void* __restrict__ Cout, const float* __restrict__ Xres,
    const float* __restrict__ gamma,
    int M, int N, int K, int lda, int ldb, int ldc,
    size_t strideA, size_t strideB, size_t strideC)
{
    (void)M; (void)N;
    __shared__ __align__(16) __bf16 As[2][BLK_M][LDS_K];   // M x K tiles (double buffer)
    __shared__ __align__(16) __bf16 Bs[2][BLK_N][LDS_K];   // stored N x K (double buffer)

    const int t     = threadIdx.x;
    const int lane  = t & 31;
    const int wave  = t >> 5;
    const int wr    = wave & 3;    // wave row: 4 x 32 = 128 M
    const int wc    = wave >> 2;   // wave col: 2 x 32 = 64 N
    const int l15   = lane & 15;
    const int khalf = lane >> 4;   // 0: lanes 0-15, 1: lanes 16-31

    const size_t z = blockIdx.z;
    const __bf16* Ab = A + z * strideA;
    const __bf16* Bb = B + z * strideB;

    const int m0 = blockIdx.y * BLK_M;
    const int n0 = blockIdx.x * BLK_N;

    // Per-thread staging coordinates.
    const int arow = t >> 1;            // 0..127
    const int akk  = (t & 1) * 16;      // 0 or 16
    const int brow = t >> 2;            // 0..63
    const int bkk  = (t & 3) * 8;       // 0,8,16,24

    // Per-tile async instruction count issued by each thread.
    constexpr int PEND = B_TRANS ? 3 : 2;

    auto issue_tile = [&](int buf, int k0) {
        const __bf16* ga = Ab + (size_t)(m0 + arow) * lda + (k0 + akk);
        async_copy_b128(ga,     &As[buf][arow][akk]);
        async_copy_b128(ga + 8, &As[buf][arow][akk + 8]);
        if (B_TRANS) {
            const __bf16* gb = Bb + (size_t)(n0 + brow) * ldb + (k0 + bkk);
            async_copy_b128(gb, &Bs[buf][brow][bkk]);
        } else {
            #pragma unroll
            for (int j = 0; j < 8; ++j)
                Bs[buf][brow][bkk + j] = Bb[(size_t)(k0 + bkk + j) * ldb + (n0 + brow)];
        }
    };

    v8f acc[2][2] = {};

    const int ntiles = K / BLK_K;
    issue_tile(0, 0);                                   // prologue fill

    for (int i = 0; i < ntiles; ++i) {
        const int buf = i & 1;
        const int k0  = i * BLK_K;

        if (i + 1 < ntiles) {
            issue_tile(buf ^ 1, k0 + BLK_K);            // overlap next tile fill
            // In-order async completion: <= PEND outstanding means the
            // current tile's loads have landed in LDS.
            asm volatile("s_wait_asynccnt %0" :: "i"(PEND) : "memory");
            if (i + 2 < ntiles)
                __builtin_prefetch(Ab + (size_t)(m0 + arow) * lda
                                      + (k0 + 2 * BLK_K + akk), 0, 1);
        } else {
            asm volatile("s_wait_asynccnt 0" ::: "memory");
        }
        __syncthreads();

        // ---- per-wave fragment loads (ISA 7.12.2 16-bit layouts) ----
        FragBF16 fa[2], fb[2];
        #pragma unroll
        for (int mt = 0; mt < 2; ++mt) {
            const __bf16* arw = &As[buf][wr * 32 + mt * 16 + l15][0];
            fa[mt].q[0] = *reinterpret_cast<const uint4*>(arw + khalf * 8);       // K = kh*8 + 0..7
            fa[mt].q[1] = *reinterpret_cast<const uint4*>(arw + 16 + khalf * 8);  // K = 16 + kh*8 + 0..7
        }
        #pragma unroll
        for (int nt = 0; nt < 2; ++nt) {
            const __bf16* brw = &Bs[buf][wc * 32 + nt * 16 + l15][0];
            fb[nt].q[0] = *reinterpret_cast<const uint4*>(brw + khalf * 8);       // K = kh*8 + 0..7
            fb[nt].q[1] = *reinterpret_cast<const uint4*>(brw + 16 + khalf * 8);  // K = 16 + kh*8 + 0..7
        }

        #pragma unroll
        for (int mt = 0; mt < 2; ++mt)
            #pragma unroll
            for (int nt = 0; nt < 2; ++nt)
                acc[mt][nt] = __builtin_amdgcn_wmma_f32_16x16x32_bf16(
                    false, fa[mt].v, false, fb[nt].v,
                    (short)0, acc[mt][nt], false, false);
        __syncthreads();   // all waves done reading buf before it is refilled
    }

    // ---- epilogue: C/D layout -> element r is (m = kh*8 + r, n = lane&15) ----
    const size_t zc = z * strideC;
    #pragma unroll
    for (int mt = 0; mt < 2; ++mt) {
        #pragma unroll
        for (int nt = 0; nt < 2; ++nt) {
            const int n = n0 + wc * 32 + nt * 16 + l15;
            #pragma unroll
            for (int r = 0; r < 8; ++r) {
                const int m = m0 + wr * 32 + mt * 16 + khalf * 8 + r;
                const size_t idx = zc + (size_t)m * ldc + n;
                const float v = acc[mt][nt][r];
                if (EPI == 0) {
                    ((float*)Cout)[idx] = v;
                } else if (EPI == 1) {
                    ((__bf16*)Cout)[idx] = (__bf16)v;
                } else {
                    ((float*)Cout)[idx] = gamma[0] * v + Xres[idx];
                }
            }
        }
    }
}

// Row softmax over contiguous rows of scoresT (length S = 1024), bf16 out.
__global__ __launch_bounds__(256) void softmax_rows_kernel(
    const float* __restrict__ scoresT, __bf16* __restrict__ betaT, int S)
{
    __shared__ float red[256];
    const size_t row = blockIdx.x;
    const float* p = scoresT + row * (size_t)S;
    __bf16*      q = betaT   + row * (size_t)S;
    const int t = threadIdx.x;

    float vmax = -3.4e38f;
    for (int i = t; i < S; i += 256) vmax = fmaxf(vmax, p[i]);
    red[t] = vmax; __syncthreads();
    for (int s = 128; s > 0; s >>= 1) {
        if (t < s) red[t] = fmaxf(red[t], red[t + s]);
        __syncthreads();
    }
    vmax = red[0]; __syncthreads();

    float e[4]; float sum = 0.f;          // S == 1024, 4 elems / thread
    #pragma unroll
    for (int j = 0; j < 4; ++j) { e[j] = __expf(p[t + j * 256] - vmax); sum += e[j]; }
    red[t] = sum; __syncthreads();
    for (int s = 128; s > 0; s >>= 1) {
        if (t < s) red[t] += red[t + s];
        __syncthreads();
    }
    const float inv = 1.0f / red[0];
    #pragma unroll
    for (int j = 0; j < 4; ++j) q[t + j * 256] = (__bf16)(e[j] * inv);
}

extern "C" void kernel_launch(void* const* d_in, const int* in_sizes, int n_in,
                              void* d_out, int out_size, void* d_ws, size_t ws_size,
                              hipStream_t stream) {
    (void)in_sizes; (void)n_in; (void)out_size; (void)ws_size;
    const int Bb = 32, S = 1024, C = 1024;

    const float* x     = (const float*)d_in[0];
    const float* Wq    = (const float*)d_in[1];
    const float* Wk    = (const float*)d_in[2];
    const float* Wv    = (const float*)d_in[3];
    const float* gamma = (const float*)d_in[4];

    char* w = (char*)d_ws;
    auto take = [&](size_t bytes) -> char* {
        char* p = w; w += (bytes + 255) & ~(size_t)255; return p;
    };
    __bf16* xb    = (__bf16*)take((size_t)Bb * S * C * 2);
    __bf16* wqb   = (__bf16*)take((size_t)C * C * 2);
    __bf16* wkb   = (__bf16*)take((size_t)C * C * 2);
    __bf16* wvb   = (__bf16*)take((size_t)C * C * 2);
    __bf16* fb    = (__bf16*)take((size_t)Bb * S * C * 2);
    __bf16* gb    = (__bf16*)take((size_t)Bb * S * C * 2);
    __bf16* hb    = (__bf16*)take((size_t)Bb * S * C * 2);
    float*  scT   = (float*) take((size_t)Bb * S * S * 4);
    __bf16* betaT = (__bf16*)take((size_t)Bb * S * S * 2);

    // 1) fp32 -> bf16 conversions
    f32_to_bf16_kernel<<<4096, 256, 0, stream>>>(x,  xb,  (size_t)Bb * S * C);
    f32_to_bf16_kernel<<<1024, 256, 0, stream>>>(Wq, wqb, (size_t)C * C);
    f32_to_bf16_kernel<<<1024, 256, 0, stream>>>(Wk, wkb, (size_t)C * C);
    f32_to_bf16_kernel<<<1024, 256, 0, stream>>>(Wv, wvb, (size_t)C * C);

    const dim3 blk(256);

    // 2) projections: [B*S, C] = [B*S, C] @ W^T   (W is N x K row-major)
    const dim3 g1(C / BLK_N, (Bb * S) / BLK_M, 1);
    gemm_wmma_bf16<true, 1><<<g1, blk, 0, stream>>>(
        xb, wqb, fb, nullptr, nullptr, Bb * S, C, C, C, C, C, 0, 0, 0);
    gemm_wmma_bf16<true, 1><<<g1, blk, 0, stream>>>(
        xb, wkb, gb, nullptr, nullptr, Bb * S, C, C, C, C, C, 0, 0, 0);
    gemm_wmma_bf16<true, 1><<<g1, blk, 0, stream>>>(
        xb, wvb, hb, nullptr, nullptr, Bb * S, C, C, C, C, C, 0, 0, 0);

    // 3) scoresT[b,t,s] = sum_c g[b,t,c] * f[b,s,c]   (A = g, B = f "transposed")
    const dim3 g2(S / BLK_N, S / BLK_M, Bb);
    gemm_wmma_bf16<true, 0><<<g2, blk, 0, stream>>>(
        gb, fb, scT, nullptr, nullptr, S, S, C, C, C, S,
        (size_t)S * C, (size_t)S * C, (size_t)S * S);

    // 4) softmax over the query axis == row softmax of scoresT
    softmax_rows_kernel<<<Bb * S, 256, 0, stream>>>(scT, betaT, S);

    // 5) out[b,t,c] = gamma * sum_s betaT[b,t,s] * h[b,s,c] + x[b,t,c]
    const dim3 g3(C / BLK_N, S / BLK_M, Bb);
    gemm_wmma_bf16<false, 2><<<g3, blk, 0, stream>>>(
        betaT, hb, d_out, x, gamma, S, C, S, S, C, C,
        (size_t)S * S, (size_t)S * C, (size_t)S * C);
}
